// OriModel_44951127720082
// MI455X (gfx1250) — matrix operand order repaired and checked by reading, not compiled, not run on hardware
//
#include <hip/hip_runtime.h>
#include <hip/hip_bf16.h>

// Shapes (fixed by the reference)
#define Bv 2
#define Dv 8
#define Nv 40
#define Hv 128
#define HID 512
#define NN (Nv * Nv)               // 1600
#define MTILE 64                   // rows per block in the big GEMM
#define TILES_PER_BI (NN / MTILE)  // 25
#define NWAVES 16                  // 512 threads

typedef __attribute__((ext_vector_type(16))) __bf16 v16bf;
typedef __attribute__((ext_vector_type(8)))  float  v8f;

struct U8x32 { uint4 lo, hi; };    // 32-byte carrier for bit-casting to v16bf

__device__ __forceinline__ unsigned bf16pk(float a, float b) {
    // round-to-nearest-even f32->bf16, packed (a low 16, b high 16)
    unsigned ua = __float_as_uint(a);
    ua += 0x7FFFu + ((ua >> 16) & 1u);
    unsigned ub = __float_as_uint(b);
    ub += 0x7FFFu + ((ub >> 16) & 1u);
    return (ua >> 16) | (ub & 0xFFFF0000u);
}

__device__ __forceinline__ unsigned short f32_to_bf16(float a) {
    unsigned ua = __float_as_uint(a);
    ua += 0x7FFFu + ((ua >> 16) & 1u);
    return (unsigned short)(ua >> 16);
}

// ---------------------------------------------------------------------------
// Prep 1: nf = max_d node_feature;  hn = nf @ W1n   -> ws  [B*N, 512] f32
// ---------------------------------------------------------------------------
__global__ __launch_bounds__(256) void prep_hn(const float* __restrict__ node,
                                               const float* __restrict__ W1n,
                                               float* __restrict__ hn) {
    const int bi = blockIdx.x;            // b*N + i
    const int b = bi / Nv, i = bi % Nv;
    __shared__ float nf[Hv];
    const int t = threadIdx.x;
    if (t < Hv) {
        float m = -3.4e38f;
        #pragma unroll
        for (int d = 0; d < Dv; ++d)
            m = fmaxf(m, node[(((size_t)b * Dv + d) * Nv + i) * Hv + t]);
        nf[t] = m;
    }
    __syncthreads();
    for (int o = t; o < HID; o += 256) {
        float s = 0.f;
        #pragma unroll 8
        for (int k = 0; k < Hv; ++k) s = fmaf(nf[k], W1n[(size_t)k * HID + o], s);
        hn[(size_t)bi * HID + o] = s;
    }
}

// ---------------------------------------------------------------------------
// Prep 2: pf = max_d pairwise;  hp = pf @ W1p   -> ws  [B*N*N, 512] f32
// ---------------------------------------------------------------------------
__global__ __launch_bounds__(256) void prep_hp(const float* __restrict__ pw,
                                               const float* __restrict__ W1p,
                                               float* __restrict__ hp) {
    const int bj = blockIdx.x;            // b*NN + jk
    const int b = bj / NN, jk = bj % NN;
    __shared__ float pf[Hv];
    const int t = threadIdx.x;
    if (t < Hv) {
        float m = -3.4e38f;
        #pragma unroll
        for (int d = 0; d < Dv; ++d)
            m = fmaxf(m, pw[(((size_t)b * Dv + d) * NN + jk) * Hv + t]);
        pf[t] = m;
    }
    __syncthreads();
    for (int o = t; o < HID; o += 256) {
        float s = 0.f;
        #pragma unroll 8
        for (int k = 0; k < Hv; ++k) s = fmaf(pf[k], W1p[(size_t)k * HID + o], s);
        hp[(size_t)bj * HID + o] = s;
    }
}

// ---------------------------------------------------------------------------
// Prep 3: W2t[col][k] = bf16(W2[k][col])  (WMMA-B friendly: contiguous in K)
// ---------------------------------------------------------------------------
__global__ __launch_bounds__(256) void w2cvt(const float* __restrict__ W2,
                                             unsigned short* __restrict__ W2t) {
    const int n = blockIdx.x;             // output column
    for (int k = threadIdx.x; k < HID; k += 256)
        W2t[(size_t)n * HID + k] = f32_to_bf16(W2[(size_t)k * HID + n]);
}

// ---------------------------------------------------------------------------
// Main fused kernel: per block = one (b,i) and a 64-row jk tile, 16 waves.
//   A[r][c] = relu(hn[b,i,c] + hp[b,jk,c] + b1[c])   (bf16, LDS)
//   h2 = relu(A @ W2 + b2)   via v_wmma_f32_16x16x32_bf16
//   out = sigmoid(scale*(h2 @ W3 + b3) + bias)       (fused epilogue)
// Wave w owns cols [w*32, w*32+32) (2 col-tiles) x 4 row-subtiles.
// BOTH operands are double-buffered: K-step ks+1's LDS (A) and L2 (B) loads
// are issued before ks's fragments are consumed, so neither ds nor load
// latency sits in front of a dependent WMMA.
// ---------------------------------------------------------------------------
#define ROWSTRIDE_U 260   // 512 bf16 = 256 uints, +4 uints pad -> 1040 B row

__global__ __launch_bounds__(512) void fused_mlp(
    const float* __restrict__ hn, const float* __restrict__ hp,
    const unsigned short* __restrict__ W2t,
    const float* __restrict__ b1, const float* __restrict__ b2,
    const float* __restrict__ W3, const float* __restrict__ b3,
    const float* __restrict__ scale, const float* __restrict__ bias,
    float* __restrict__ out) {

    __shared__ unsigned Atile[MTILE * ROWSTRIDE_U];   // 66,560 B packed bf16
    __shared__ float red[NWAVES][MTILE];              // per-wave row partials

    const int blk  = blockIdx.x;
    const int bi   = blk / TILES_PER_BI;              // b*N + i
    const int tile = blk % TILES_PER_BI;
    const int b    = bi / Nv;
    const int jkbase = tile * MTILE;

    const int t    = threadIdx.x;
    const int w    = t >> 5;                          // wave id, 0..15
    const int lane = t & 31;
    const int nlo  = lane & 15;
    const int half = lane >> 4;

    // ---- Stage 1: build the bf16 activation tile in LDS -------------------
    {
        const int r  = t >> 3;                        // 8 threads per row
        const int c0 = (t & 7) * 64;                  // 64 contiguous cols
        const float* hpr = hp + ((size_t)(b * NN + jkbase + r)) * HID + c0;
        const float* hnr = hn + (size_t)bi * HID + c0;
        const float* b1r = b1 + c0;
        unsigned* dst = &Atile[r * ROWSTRIDE_U + (c0 >> 1)];
        #pragma unroll
        for (int ii = 0; ii < 16; ++ii) {
            const float4 hv = *(const float4*)(hpr + ii * 4);
            const float4 nv = *(const float4*)(hnr + ii * 4);
            const float4 bv = *(const float4*)(b1r + ii * 4);
            float x0 = fmaxf(hv.x + nv.x + bv.x, 0.f);
            float x1 = fmaxf(hv.y + nv.y + bv.y, 0.f);
            float x2 = fmaxf(hv.z + nv.z + bv.z, 0.f);
            float x3 = fmaxf(hv.w + nv.w + bv.w, 0.f);
            dst[ii * 2 + 0] = bf16pk(x0, x1);
            dst[ii * 2 + 1] = bf16pk(x2, x3);
        }
    }
    __syncthreads();

    // ---- Stage 2: WMMA K-loop, A and B double-buffered --------------------
    v8f acc[4][2] = {};
    float b2l[2], w3l[2];
    #pragma unroll
    for (int ct = 0; ct < 2; ++ct) {
        const int col = w * 32 + ct * 16 + nlo;
        b2l[ct] = b2[col];
        w3l[ct] = W3[col];
    }

    // B fragment base (ISA B layout: n=l&15, K=ks*32+half*8+0..7, +16 for
    // elements 8..15 -> two 16B loads from the K-major bf16 W2).
    const unsigned short* W2base =
        W2t + ((size_t)(w * 32 + nlo) * HID + half * 8);
    // A fragment base in uints (ISA A layout: m=l&15; g0..3 K=ks*32+half*8+0..7,
    // g4..7 at +16 -> two 16B LDS loads, bank-conflict-free row stride).
    const int Abase = nlo * ROWSTRIDE_U + half * 4;

    U8x32 aa[4], an[4], bb[2], bn[2];
    #pragma unroll
    for (int rt = 0; rt < 4; ++rt) {
        const int base = Abase + rt * 16 * ROWSTRIDE_U;
        aa[rt].lo = *(const uint4*)&Atile[base];
        aa[rt].hi = *(const uint4*)&Atile[base + 8];
    }
    #pragma unroll
    for (int ct = 0; ct < 2; ++ct) {
        const uint4* p = (const uint4*)(W2base + ct * 16 * HID);
        bb[ct].lo = p[0];
        bb[ct].hi = p[2];
    }

    #pragma unroll
    for (int ks = 0; ks < 16; ++ks) {
        // issue next K-step's loads before consuming this step's fragments
        if (ks < 15) {
            #pragma unroll
            for (int rt = 0; rt < 4; ++rt) {
                const int base =
                    Abase + rt * 16 * ROWSTRIDE_U + (ks + 1) * 16;
                an[rt].lo = *(const uint4*)&Atile[base];
                an[rt].hi = *(const uint4*)&Atile[base + 8];
            }
            #pragma unroll
            for (int ct = 0; ct < 2; ++ct) {
                const uint4* p =
                    (const uint4*)(W2base + ct * 16 * HID + (ks + 1) * 32);
                bn[ct].lo = p[0];
                bn[ct].hi = p[2];
            }
        }
        #pragma unroll
        for (int ct = 0; ct < 2; ++ct) {
            const v16bf bfr = __builtin_bit_cast(v16bf, bb[ct]);
            #pragma unroll
            for (int rt = 0; rt < 4; ++rt) {
                const v16bf afr = __builtin_bit_cast(v16bf, aa[rt]);
                acc[rt][ct] = __builtin_amdgcn_wmma_f32_16x16x32_bf16(
                    false, afr, false, bfr, (short)0, acc[rt][ct],
                    false, false);
            }
        }
        #pragma unroll
        for (int rt = 0; rt < 4; ++rt) aa[rt] = an[rt];
        #pragma unroll
        for (int ct = 0; ct < 2; ++ct) bb[ct] = bn[ct];
    }

    // ---- Stage 3: fused epilogue ------------------------------------------
    // C layout: VGPR v of a 16x16 tile holds row M = v + 8*half, col N = l&15.
    #pragma unroll
    for (int rt = 0; rt < 4; ++rt) {
        #pragma unroll
        for (int v = 0; v < 8; ++v) {
            float p = 0.f;
            #pragma unroll
            for (int ct = 0; ct < 2; ++ct)
                p = fmaf(fmaxf(acc[rt][ct][v] + b2l[ct], 0.f), w3l[ct], p);
            // reduce the 16 lanes (same half) holding this row's 16 columns
            p += __shfl_xor(p, 1);
            p += __shfl_xor(p, 2);
            p += __shfl_xor(p, 4);
            p += __shfl_xor(p, 8);
            if (nlo == 0) red[w][rt * 16 + half * 8 + v] = p;
        }
    }
    __syncthreads();

    if (t < MTILE) {
        float s = 0.f;
        #pragma unroll
        for (int ww = 0; ww < NWAVES; ++ww) s += red[ww][t];
        const float vv = s + b3[0];
        const float z  = scale[0] * vv + bias[0];
        out[(size_t)bi * NN + jkbase + t] = 1.f / (1.f + __expf(-z));
    }
}

// ---------------------------------------------------------------------------
extern "C" void kernel_launch(void* const* d_in, const int* in_sizes, int n_in,
                              void* d_out, int out_size, void* d_ws, size_t ws_size,
                              hipStream_t stream) {
    const float* node  = (const float*)d_in[0];
    const float* pw    = (const float*)d_in[1];
    const float* W1n   = (const float*)d_in[2];
    const float* W1p   = (const float*)d_in[3];
    const float* b1    = (const float*)d_in[4];
    const float* W2    = (const float*)d_in[5];
    const float* b2    = (const float*)d_in[6];
    const float* W3    = (const float*)d_in[7];
    const float* b3    = (const float*)d_in[8];
    const float* scale = (const float*)d_in[9];
    const float* bias  = (const float*)d_in[10];
    float* out = (float*)d_out;

    char* ws = (char*)d_ws;
    float* hn = (float*)ws;                                   // 2*40*512 f32
    float* hp = (float*)(ws + (size_t)Bv * Nv * HID * 4);     // 2*1600*512 f32
    unsigned short* W2t =
        (unsigned short*)(ws + (size_t)Bv * Nv * HID * 4
                             + (size_t)Bv * NN * HID * 4);    // 512*512 bf16

    prep_hn<<<Bv * Nv, 256, 0, stream>>>(node, W1n, hn);
    prep_hp<<<Bv * NN, 256, 0, stream>>>(pw, W1p, hp);
    w2cvt<<<HID, 256, 0, stream>>>(W2, W2t);
    fused_mlp<<<Bv * Nv * TILES_PER_BI, 512, 0, stream>>>(
        hn, hp, W2t, b1, b2, W3, b3, scale, bias, out);
}
